// HKSU_11433202942762
// MI455X (gfx1250) — matrix operand order repaired and checked by reading, not compiled, not run on hardware
//
#include <hip/hip_runtime.h>
#include <stdint.h>

// ---------------- problem constants ----------------
#define B_   8
#define T_   128
#define D_   512
#define M_   256
#define K8   8      // TOPK
#define NH_  2
#define RTOT (B_*T_)   // 1024 rows

typedef float v8f __attribute__((ext_vector_type(8)));
typedef __bf16 v8bf __attribute__((ext_vector_type(8)));
typedef __bf16 v16bf __attribute__((ext_vector_type(16)));

static __device__ inline __bf16 f2bf(float f) {
  unsigned u = __builtin_bit_cast(unsigned, f);
  // round-to-nearest-even truncation to bf16
  unsigned r = (u + 0x7FFFu + ((u >> 16) & 1u)) >> 16;
  unsigned short s = (unsigned short)r;
  return __builtin_bit_cast(__bf16, s);
}

static __device__ inline float fast_rcp(float x) { return __builtin_amdgcn_rcpf(x); }
static __device__ inline float fast_sigmoid(float x) {
  return fast_rcp(1.f + __expf(-x));
}

static __device__ inline float wred(float v) {
#pragma unroll
  for (int o = 16; o > 0; o >>= 1) v += __shfl_xor(v, o, 32);
  return v;
}

// =====================================================================
// f32 -> bf16 staging conversion (grid-stride, float4-vectorized)
// =====================================================================
__global__ __launch_bounds__(256) void cvt_bf16(
    const float* __restrict__ in, __bf16* __restrict__ out, int n4)
{
  int i = blockIdx.x * 256 + threadIdx.x;
  const int stride = gridDim.x * 256;
  for (; i < n4; i += stride) {
    float4 v = ((const float4*)in)[i];
    v8bf dummy; (void)dummy;
    __bf16 o0 = f2bf(v.x), o1 = f2bf(v.y), o2 = f2bf(v.z), o3 = f2bf(v.w);
    typedef __bf16 v4bf __attribute__((ext_vector_type(4)));
    v4bf o; o[0] = o0; o[1] = o1; o[2] = o2; o[3] = o3;
    ((v4bf*)out)[i] = o;
  }
}

// ---- A 16x32 bf16 fragment: lanes 0-15 hold K=0..15;
//      lanes 16-31 hold K=16..31 (contiguous, matches B)
static __device__ inline v16bf load_fragA(const __bf16* __restrict__ p, int half) {
  return *(const v16bf*)(p + half * 16);
}

// ---- B 32x16 bf16 fragment (contiguous k order, same as A): lanes 0-15 hold K=0..15,
//      lanes 16-31 hold K=16..31, column n = lane&15  (contiguous 32B load)
static __device__ inline v16bf load_fragB(const __bf16* __restrict__ p, int half) {
  return *(const v16bf*)(p + half * 16);
}

// =====================================================================
// Y[M x N] = X[M x K] @ W[N x K]^T   (bf16 operands, f32 accumulate)
// one wave computes a 16 x (NT*16) output strip: A loaded once per
// k-step, reused across NT B-tiles / accumulators.
// grid = (M/16, N/(NT*16)), block = 32
// optional epilogue: Y = sigmoid(gate) * Y   (gate same shape as Y)
// =====================================================================
template <int NT>
__global__ __launch_bounds__(32) void wmma_gemm(
    const __bf16* __restrict__ X, const __bf16* __restrict__ W,
    float* __restrict__ Y, int N, int K, const float* __restrict__ gate)
{
  const int tileM = blockIdx.x, tileN0 = blockIdx.y * NT;
  const int lane = threadIdx.x;
  const int rc = lane & 15, half = lane >> 4;

  const __bf16* xrow = X + (size_t)(tileM * 16 + rc) * K;
  const __bf16* wrow[NT];
#pragma unroll
  for (int nt = 0; nt < NT; ++nt)
    wrow[nt] = W + (size_t)((tileN0 + nt) * 16 + rc) * K;

  v8f acc[NT];
#pragma unroll
  for (int nt = 0; nt < NT; ++nt) acc[nt] = (v8f){};

#pragma unroll 4
  for (int k0 = 0; k0 < K; k0 += 32) {
    v16bf A = load_fragA(xrow + k0, half);
#pragma unroll
    for (int nt = 0; nt < NT; ++nt) {
      v16bf Bm = load_fragB(wrow[nt] + k0, half);
      acc[nt] = __builtin_amdgcn_wmma_f32_16x16x32_bf16(
          /*neg_a=*/false, A, /*neg_b=*/false, Bm,
          /*c_mod=*/(short)0, acc[nt], /*reuse_a=*/false, /*reuse_b=*/false);
    }
  }

  // D layout: lane covers column n = rc; VGPR r holds row m = r + 8*half
  if (gate != nullptr) {
#pragma unroll
    for (int nt = 0; nt < NT; ++nt) {
      const int n = (tileN0 + nt) * 16 + rc;
#pragma unroll
      for (int r = 0; r < 8; ++r) {
        const int m = tileM * 16 + r + 8 * half;
        Y[(size_t)m * N + n] = acc[nt][r] * fast_sigmoid(gate[(size_t)m * N + n]);
      }
    }
  } else {
#pragma unroll
    for (int nt = 0; nt < NT; ++nt) {
      const int n = (tileN0 + nt) * 16 + rc;
#pragma unroll
      for (int r = 0; r < 8; ++r) {
        const int m = tileM * 16 + r + 8 * half;
        Y[(size_t)m * N + n] = acc[nt][r];
      }
    }
  }
}

// =====================================================================
// Bpre[j, row] = h[row,:] . hh_bw[j,:] + hh_bb[j]
// grid = (RTOT, NH), block = 32
// =====================================================================
__global__ __launch_bounds__(32) void rowdot(
    const float* __restrict__ X, const float* __restrict__ w,
    const float* __restrict__ bb, float* __restrict__ out)
{
  const int row = blockIdx.x, j = blockIdx.y, lane = threadIdx.x;
  const float* x = X + (size_t)row * D_;
  const float* ww = w + (size_t)j * D_;
  float s = 0.f;
  for (int i = lane; i < D_; i += 32) s += x[i] * ww[i];
  s = wred(s);
  if (lane == 0) out[j * RTOT + row] = s + bb[j];
}

// =====================================================================
// kron address + top-8 (descending, ties -> lower index like lax.top_k)
// executed cooperatively by all 256 threads
// =====================================================================
static __device__ void addr_topk(const float* __restrict__ zsrc, float inv_tau,
                                 int tid, float* zrow, float* pe, float* pp,
                                 unsigned* akey, unsigned* rk, unsigned* ri,
                                 int* oidx, float* ow)
{
  if (tid < 32) zrow[tid] = zsrc[tid];
  __syncthreads();
  if (tid < 32) {
    int g = tid & ~15;
    float mx = -3.4e38f;
    for (int i = 0; i < 16; ++i) mx = fmaxf(mx, zrow[g + i]);
    pe[tid] = __expf((zrow[tid] - mx) * inv_tau);
  }
  __syncthreads();
  if (tid < 32) {
    int g = tid & ~15;
    float s = 0.f;
    for (int i = 0; i < 16; ++i) s += pe[g + i];
    pp[tid] = pe[tid] * fast_rcp(s);
  }
  __syncthreads();
  // addr[i*16+j] = p0[i] * p1[j]; positive floats -> uint compare == float compare
  float val = pp[tid >> 4] * pp[16 + (tid & 15)];
  akey[tid] = __float_as_uint(val);
  __syncthreads();
  for (int pass = 0; pass < 8; ++pass) {
    rk[tid] = akey[tid]; ri[tid] = (unsigned)tid;
    __syncthreads();
    for (int s = 128; s >= 1; s >>= 1) {
      if (tid < s) {
        unsigned ka = rk[tid], kb = rk[tid + s];
        unsigned ia = ri[tid], ib = ri[tid + s];
        if (kb > ka || (kb == ka && ib < ia)) { rk[tid] = kb; ri[tid] = ib; }
      }
      __syncthreads();
    }
    if (tid == 0) {
      int bi = (int)ri[0];
      oidx[pass] = bi;
      ow[pass] = __uint_as_float(rk[0]);
      akey[bi] = 0u;
    }
    __syncthreads();
  }
}

// =====================================================================
// Sequential scan: one workgroup per batch element; 8 waves = 8 slots,
// each lane owns 16 contiguous d-elements.
// =====================================================================
__global__ __launch_bounds__(256) void scan_kernel(
    const float* __restrict__ ZK, const float* __restrict__ ZQ,
    const float* __restrict__ Kp, const float* __restrict__ VKp,
    const float* __restrict__ VVp, const float* __restrict__ Bpre,
    const float* __restrict__ h,
    const float* __restrict__ log_tau_w, const float* __restrict__ log_tau_r,
    const float* __restrict__ gamma,
    const float* __restrict__ zK_in, const float* __restrict__ zV_in,
    float* __restrict__ Ks, float* __restrict__ Vs,            // in d_out, pre-seeded
    float* __restrict__ zK_out, float* __restrict__ zV_out,
    float* __restrict__ widx_out, float* __restrict__ ridx_out,
    float* __restrict__ RO)
{
  __shared__ float zrow[32], pe[32], pp[32];
  __shared__ unsigned akey[256], rk[256], ri[256];
  __shared__ int   widx_s[8]; __shared__ float ww_s[8];
  __shared__ int   ridx_s[8]; __shared__ float rw_s[8];
  __shared__ float zk_l[M_], zv_l[M_];
  __shared__ float kmir[8][D_], vmir[8][D_];
  __shared__ float rel_s[8];
  __shared__ float ro_s[D_];

  const int b = blockIdx.x, tid = threadIdx.x;
  const int wv = tid >> 5, lane = tid & 31, dbase = lane * 16;
  const float itw = __expf(-log_tau_w[0]);
  const float itr = __expf(-log_tau_r[0]);
  const float spg = log1pf(__expf(gamma[0]));
  const float scale = rsqrtf((float)D_);

  zk_l[tid] = zK_in[b * M_ + tid];
  zv_l[tid] = zV_in[b * M_ + tid];
  __syncthreads();

  for (int t = 0; t < T_; ++t) {
    const int row = b * T_ + t;

    // ---------- addresses ----------
    addr_topk(ZK + (size_t)row * 32, itw, tid, zrow, pe, pp, akey, rk, ri, widx_s, ww_s);
    addr_topk(ZQ + (size_t)row * 32, itr, tid, zrow, pe, pp, akey, rk, ri, ridx_s, rw_s);

    // ---------- write-slot update (wave wv owns slot widx_s[wv]) ----------
    const int   s  = widx_s[wv];
    const float w  = ww_s[wv];
    const float dec = powf(1.f - w, spg);
    if (lane == 0) {
      zk_l[s] = dec * zk_l[s] + w;
      zv_l[s] = dec * zv_l[s] + w;
    }
    float* Kr = Ks + ((size_t)b * M_ + s) * D_;
    float* Vr = Vs + ((size_t)b * M_ + s) * D_;
    float ka[16], va[16];
#pragma unroll
    for (int i = 0; i < 16; ++i) { ka[i] = Kr[dbase + i] * dec; va[i] = Vr[dbase + i] * dec; }

#pragma unroll
    for (int j = 0; j < NH_; ++j) {
      const float* kp = Kp  + ((size_t)j * RTOT + row) * D_ + dbase;
      const float* vk = VKp + ((size_t)j * RTOT + row) * D_ + dbase;
      const float* vv = VVp + ((size_t)j * RTOT + row) * D_ + dbase;
      float kj[16]; float ss = 0.f;
#pragma unroll
      for (int i = 0; i < 16; ++i) {
        float x = kp[i];
        float sj = x * fast_sigmoid(x);   // silu
        kj[i] = sj; ss += sj * sj;
      }
      ss = wred(ss);
      float inv  = fast_rcp(fmaxf(sqrtf(ss), 1e-12f));
      float beta = 2.f * fast_sigmoid(Bpre[j * RTOT + row]);
      float dk = 0.f, dv = 0.f;
#pragma unroll
      for (int i = 0; i < 16; ++i) { dk += kj[i] * ka[i]; dv += kj[i] * va[i]; }
      dk = wred(dk); dv = wred(dv);
      const float cK = beta * inv * inv * dk;  // beta * kh * (kh . K)
      const float cV = beta * inv * inv * dv;
#pragma unroll
      for (int i = 0; i < 16; ++i) {
        float k_ = kj[i];
        ka[i] += k_ * (beta * vk[i] - cK);
        va[i] += k_ * (beta * vv[i] - cV);
      }
    }
#pragma unroll
    for (int i = 0; i < 16; ++i) {
      Kr[dbase + i] = ka[i]; kmir[wv][dbase + i] = ka[i];
      Vr[dbase + i] = va[i]; vmir[wv][dbase + i] = va[i];
    }
    __threadfence_block();
    __syncthreads();

    // ---------- read / attention ----------
    const int rsl = ridx_s[wv];
    int wm = -1;
#pragma unroll
    for (int m2 = 0; m2 < 8; ++m2) if (widx_s[m2] == rsl) wm = m2;
    float kr[16], vr[16];
    if (wm >= 0) {
#pragma unroll
      for (int i = 0; i < 16; ++i) { kr[i] = kmir[wm][dbase + i]; vr[i] = vmir[wm][dbase + i]; }
    } else {
      const float* Kr2 = Ks + ((size_t)b * M_ + rsl) * D_;
      const float* Vr2 = Vs + ((size_t)b * M_ + rsl) * D_;
#pragma unroll
      for (int i = 0; i < 16; ++i) { kr[i] = Kr2[dbase + i]; vr[i] = Vr2[dbase + i]; }
    }
    const float zkr = zk_l[rsl], zvr = zv_l[rsl];
    const float* hrow = h + (size_t)row * D_ + dbase;
    float rl = 0.f;
#pragma unroll
    for (int i = 0; i < 16; ++i) rl += kr[i] * hrow[i];
    rl = wred(rl);
    if (lane == 0) rel_s[wv] = rl * fast_rcp(zkr) * scale;
    ro_s[tid] = 0.f; ro_s[tid + 256] = 0.f;
    __syncthreads();

    float mx = -3.4e38f;
#pragma unroll
    for (int m2 = 0; m2 < 8; ++m2) mx = fmaxf(mx, rel_s[m2]);
    float den = 0.f;
#pragma unroll
    for (int m2 = 0; m2 < 8; ++m2) den += __expf(rel_s[m2] - mx);
    const float attn = __expf(rel_s[wv] - mx) * fast_rcp(den);
    const float izv = attn * fast_rcp(zvr);
#pragma unroll
    for (int i = 0; i < 16; ++i) atomicAdd(&ro_s[dbase + i], vr[i] * izv);
    __syncthreads();

    float* ro_g = RO + (size_t)row * D_;
    ro_g[tid] = ro_s[tid];
    ro_g[tid + 256] = ro_s[tid + 256];
    if (t == T_ - 1 && tid < 8) {
      widx_out[b * 8 + tid] = (float)widx_s[tid];
      ridx_out[b * 8 + tid] = (float)ridx_s[tid];
    }
    __syncthreads();
  }

  zK_out[b * M_ + tid] = zk_l[tid];
  zV_out[b * M_ + tid] = zv_l[tid];
}

// =====================================================================
// host-side launcher
// =====================================================================
extern "C" void kernel_launch(void* const* d_in, const int* in_sizes, int n_in,
                              void* d_out, int out_size, void* d_ws, size_t ws_size,
                              hipStream_t stream) {
  (void)in_sizes; (void)n_in; (void)out_size; (void)ws_size;

  const float* h        = (const float*)d_in[0];
  const float* K_slots  = (const float*)d_in[1];
  const float* V_slots  = (const float*)d_in[2];
  const float* z_K      = (const float*)d_in[3];
  const float* z_V      = (const float*)d_in[4];
  const float* W_k      = (const float*)d_in[5];
  const float* W_q      = (const float*)d_in[6];
  const float* ltw      = (const float*)d_in[7];
  const float* ltr      = (const float*)d_in[8];
  const float* hh_k     = (const float*)d_in[9];
  const float* hh_vk    = (const float*)d_in[10];
  const float* hh_vv    = (const float*)d_in[11];
  const float* hh_bw    = (const float*)d_in[12];
  const float* hh_bb    = (const float*)d_in[13];
  const float* gamma    = (const float*)d_in[14];
  const float* W_out    = (const float*)d_in[15];
  const float* W_gate   = (const float*)d_in[16];

  float* out = (float*)d_out;
  float* ws  = (float*)d_ws;

  // ----- output layout (flat concat, return order) -----
  const size_t OFF_OUT  = 0;                         // B*T*D
  const size_t OFF_K    = (size_t)B_ * T_ * D_;      // B*M*D
  const size_t OFF_V    = OFF_K  + (size_t)B_ * M_ * D_;
  const size_t OFF_ZK   = OFF_V  + (size_t)B_ * M_ * D_;
  const size_t OFF_ZV   = OFF_ZK + (size_t)B_ * M_;
  const size_t OFF_WIDX = OFF_ZV + (size_t)B_ * M_;
  const size_t OFF_RIDX = OFF_WIDX + (size_t)B_ * K8;

  // ----- workspace layout: f32 region -----
  const size_t KP  = 0;                              // NH * RTOT * D
  const size_t VKP = KP  + (size_t)NH_ * RTOT * D_;
  const size_t VVP = VKP + (size_t)NH_ * RTOT * D_;
  const size_t GP  = VVP + (size_t)NH_ * RTOT * D_;  // RTOT * D
  const size_t ZKp = GP  + (size_t)RTOT * D_;        // RTOT * 32
  const size_t ZQp = ZKp + (size_t)RTOT * 32;
  const size_t BP  = ZQp + (size_t)RTOT * 32;        // NH * RTOT
  const size_t ROp = BP  + (size_t)NH_ * RTOT;       // RTOT * D
  const size_t F32_END = ROp + (size_t)RTOT * D_;

  // ----- workspace layout: bf16 staging region (after f32 region) -----
  __bf16* bws = (__bf16*)(ws + F32_END);
  const size_t HB   = 0;                             // RTOT * D
  const size_t WKB  = HB   + (size_t)RTOT * D_;      // 32 * D
  const size_t WQB  = WKB  + (size_t)32 * D_;
  const size_t HKB  = WQB  + (size_t)32 * D_;        // NH * D * D
  const size_t HVKB = HKB  + (size_t)NH_ * D_ * D_;
  const size_t HVVB = HVKB + (size_t)NH_ * D_ * D_;
  const size_t WGB  = HVVB + (size_t)NH_ * D_ * D_;  // D * D
  const size_t WOB  = WGB  + (size_t)D_ * D_;        // D * D
  const size_t ROB  = WOB  + (size_t)D_ * D_;        // RTOT * D

  // seed K_s / V_s state in-place in output buffer
  hipMemcpyAsync(out + OFF_K, K_slots, (size_t)B_ * M_ * D_ * sizeof(float),
                 hipMemcpyDeviceToDevice, stream);
  hipMemcpyAsync(out + OFF_V, V_slots, (size_t)B_ * M_ * D_ * sizeof(float),
                 hipMemcpyDeviceToDevice, stream);

  // ----- stage all GEMM operands to bf16 (one-time conversion) -----
  const int cvtBlk = 256, cvtGrid = 512;
  cvt_bf16<<<cvtGrid, cvtBlk, 0, stream>>>(h,      bws + HB,   RTOT * D_ / 4);
  cvt_bf16<<<cvtGrid, cvtBlk, 0, stream>>>(W_k,    bws + WKB,  32 * D_ / 4);
  cvt_bf16<<<cvtGrid, cvtBlk, 0, stream>>>(W_q,    bws + WQB,  32 * D_ / 4);
  cvt_bf16<<<cvtGrid, cvtBlk, 0, stream>>>(hh_k,   bws + HKB,  NH_ * D_ * D_ / 4);
  cvt_bf16<<<cvtGrid, cvtBlk, 0, stream>>>(hh_vk,  bws + HVKB, NH_ * D_ * D_ / 4);
  cvt_bf16<<<cvtGrid, cvtBlk, 0, stream>>>(hh_vv,  bws + HVVB, NH_ * D_ * D_ / 4);
  cvt_bf16<<<cvtGrid, cvtBlk, 0, stream>>>(W_gate, bws + WGB,  D_ * D_ / 4);
  cvt_bf16<<<cvtGrid, cvtBlk, 0, stream>>>(W_out,  bws + WOB,  D_ * D_ / 4);

  const dim3 blk(32);
  const dim3 gridBig(RTOT / 16, D_ / 64);   // 64 x 8  (NT=4 -> 16x64 per wave)
  const dim3 gridAddr(RTOT / 16, 1);        // 64 x 1  (NT=2 -> 16x32 per wave)

  // phase 1: all h-only projections (bf16 WMMA GEMMs)
  for (int j = 0; j < NH_; ++j) {
    wmma_gemm<4><<<gridBig, blk, 0, stream>>>(bws + HB, bws + HKB  + (size_t)j * D_ * D_,
                                              ws + KP  + (size_t)j * RTOT * D_, D_, D_, nullptr);
    wmma_gemm<4><<<gridBig, blk, 0, stream>>>(bws + HB, bws + HVKB + (size_t)j * D_ * D_,
                                              ws + VKP + (size_t)j * RTOT * D_, D_, D_, nullptr);
    wmma_gemm<4><<<gridBig, blk, 0, stream>>>(bws + HB, bws + HVVB + (size_t)j * D_ * D_,
                                              ws + VVP + (size_t)j * RTOT * D_, D_, D_, nullptr);
  }
  wmma_gemm<4><<<gridBig,  blk, 0, stream>>>(bws + HB, bws + WGB, ws + GP,  D_, D_, nullptr);
  wmma_gemm<2><<<gridAddr, blk, 0, stream>>>(bws + HB, bws + WKB, ws + ZKp, 32, D_, nullptr);
  wmma_gemm<2><<<gridAddr, blk, 0, stream>>>(bws + HB, bws + WQB, ws + ZQp, 32, D_, nullptr);
  rowdot<<<dim3(RTOT, NH_), blk, 0, stream>>>(h, hh_bw, hh_bb, ws + BP);

  // phase 2: sequential scan (one workgroup per batch element)
  scan_kernel<<<B_, 256, 0, stream>>>(
      ws + ZKp, ws + ZQp, ws + KP, ws + VKP, ws + VVP, ws + BP, h,
      ltw, ltr, gamma, z_K, z_V,
      out + OFF_K, out + OFF_V, out + OFF_ZK, out + OFF_ZV,
      out + OFF_WIDX, out + OFF_RIDX, ws + ROp);

  // phase 3: out = sigmoid(h @ W_gate^T) * (readout @ W_out^T)
  cvt_bf16<<<cvtGrid, cvtBlk, 0, stream>>>(ws + ROp, bws + ROB, RTOT * D_ / 4);
  wmma_gemm<4><<<gridBig, blk, 0, stream>>>(bws + ROB, bws + WOB, out + OFF_OUT, D_, D_, ws + GP);
}
